// Decagon_23287312679606
// MI455X (gfx1250) — matrix operand order repaired and checked by reading, not compiled, not run on hardware
//
#include <hip/hip_runtime.h>

// ---------------------------------------------------------------------------
// Decagon GNN on MI455X (gfx1250, wave32).
//   phase 1: per-edge-type scatter-add of raw features (float atomics, L2)
//   phase 2: dense bf16 WMMA GEMMs (v_wmma_f32_16x16x32_bf16), weights
//            pre-packed to fragment order and staged through LDS
//   phase 3: GEMM-factored decoders + per-edge 64-wide dot products
// Layer-2 gene embeddings are dead code (outputs only use drug embeddings).
// ---------------------------------------------------------------------------

namespace {
constexpr int FIN = 128;
constexpr int DIM = 64;
constexpr int NG  = 20000;
constexpr int ND  = 4000;
}

typedef __attribute__((ext_vector_type(16))) __bf16 v16bf;
typedef __attribute__((ext_vector_type(2)))  __bf16 v2bf;
typedef __attribute__((ext_vector_type(8)))  float  v8f;

static __device__ __forceinline__ __bf16 f2bf(float f) {
  unsigned u = __builtin_bit_cast(unsigned, f);
  u += 0x7FFFu + ((u >> 16) & 1u);          // round-to-nearest-even
  unsigned short h = (unsigned short)(u >> 16);
  return __builtin_bit_cast(__bf16, h);
}

// packed f32 -> 2x bf16 (v_cvt_pk_bf16_f32 when available)
static __device__ __forceinline__ v2bf f2bf2(float lo, float hi) {
#if __has_builtin(__builtin_amdgcn_cvt_pk_bf16_f32)
  return __builtin_amdgcn_cvt_pk_bf16_f32(lo, hi);
#else
  v2bf r;
  r[0] = f2bf(lo);
  r[1] = f2bf(hi);
  return r;
#endif
}

union AFrag {
  v16bf v;
  v2bf  p[8];
};

// out[dst[e], :] += X[src[e], :]  (row length F = 2^logF floats; float4 per thread)
__global__ void scatter_add_k(const float* __restrict__ X,
                              const int* __restrict__ src,
                              const int* __restrict__ dst,
                              float* __restrict__ out,
                              long long E, int logF) {
  const int logC = logF - 2;                 // float4 chunks per row
  const long long C = 1ll << logC;
  const long long total  = E << logC;
  const long long stride = (long long)gridDim.x * blockDim.x;
  for (long long idx = (long long)blockIdx.x * blockDim.x + threadIdx.x;
       idx < total; idx += stride) {
    long long e  = idx >> logC;
    int       c4 = (int)(idx & (C - 1));
    long long en = (idx + stride) >> logC;   // prefetch upcoming edge indices
    if (en < E) {
      __builtin_prefetch(&src[en], 0, 1);
      __builtin_prefetch(&dst[en], 0, 1);
    }
    const float4 v = ((const float4*)(X + ((long long)src[e] << logF)))[c4];
    float* o = out + ((long long)dst[e] << logF) + (long long)c4 * 4;
    atomicAdd(o + 0, v.x);
    atomicAdd(o + 1, v.y);
    atomicAdd(o + 2, v.z);
    atomicAdd(o + 3, v.w);
  }
}

// Pack a K x 64 f32 weight into bf16 WMMA B-fragment order:
//   Wp[((ks*4 + tn)*32 + lane)*16 + i] = bf16( W[(ks*32 + 16*(lane>>4) + i)*64
//                                                + tn*16 + (lane&15)] )
__global__ void pack_w_k(const float* __restrict__ W, __bf16* __restrict__ Wp,
                         int K) {
  int t = blockIdx.x * blockDim.x + threadIdx.x;
  int total = (K / 32) * 4 * 32;
  if (t >= total) return;
  int lane = t & 31;
  int tn   = (t >> 5) & 3;
  int ks   = t >> 7;
  int r = lane & 15, h = lane >> 4;
  AFrag v;
#pragma unroll
  for (int i = 0; i < 8; ++i) {
    int k = ks * 32 + h * 16 + 2 * i;
    v.p[i] = f2bf2(W[(size_t)k * DIM + tn * 16 + r],
                   W[(size_t)(k + 1) * DIM + tn * 16 + r]);
  }
  *(v16bf*)(Wp + (size_t)t * 16) = v.v;
}

// C[M x 64] = act( sum_p A_p[M x K] @ B_p[K x 64] ).
// One wave computes a full 16-row x 64-col stripe (4 accumulators, A fragment
// reused across 4 WMMAs per K-step). B fragments staged per-pair in LDS.
// SCALE folds diag(prescale) @ ... @ diag(postscale) (DEDICOM decoder).
template <int K, bool SCALE>
__global__ void gemm_wmma_k(const float* A0, const float* A1, const float* A2,
                            const float* A3, const float* A4,
                            const __bf16* W0, const __bf16* W1, const __bf16* W2,
                            const __bf16* W3, const __bf16* W4,
                            int npairs, int M,
                            const float* __restrict__ prescale,
                            const float* __restrict__ postscale,
                            float* __restrict__ C, int doRelu) {
  extern __shared__ char smem_raw[];
  __bf16* smem = (__bf16*)smem_raw;

  const int lane = threadIdx.x & 31;
  const int wave = threadIdx.x >> 5;
  const int stripe = blockIdx.x * (blockDim.x >> 5) + wave;   // wave-uniform
  const bool active = stripe < (M / 16);
  const int r = lane & 15, h = lane >> 4;

  const v8f vzero = {0.f, 0.f, 0.f, 0.f, 0.f, 0.f, 0.f, 0.f};
  v8f acc[4] = {vzero, vzero, vzero, vzero};

  for (int p = 0; p < npairs; ++p) {
    const __bf16* Wp = (p == 0) ? W0 : (p == 1) ? W1 : (p == 2) ? W2
                     : (p == 3) ? W3 : W4;
    // cooperative stage of packed weights (K*64 bf16 = K*128 bytes) into LDS
    {
      const uint4* s = (const uint4*)Wp;
      uint4*       d = (uint4*)smem;
      for (int c = threadIdx.x; c < K * DIM / 8; c += blockDim.x) d[c] = s[c];
    }
    __syncthreads();

    if (active) {
      const float* A = (p == 0) ? A0 : (p == 1) ? A1 : (p == 2) ? A2
                     : (p == 3) ? A3 : A4;
      const float* Arow = A + (size_t)(stripe * 16 + r) * K;
#pragma unroll
      for (int ks = 0; ks < K / 32; ++ks) {
        const int kk = ks * 32;
        // A fragment: two contiguous 8-float runs -> 4x b128 loads
        const float4* g0 = (const float4*)(Arow + kk + h * 16);
        const float4* g1 = (const float4*)(Arow + kk + h * 16 + 8);
        float4 f0 = g0[0], f1 = g0[1], f2 = g1[0], f3 = g1[1];
        if constexpr (SCALE) {
          const float4* s0 = (const float4*)(prescale + kk + h * 16);
          const float4* s1 = (const float4*)(prescale + kk + h * 16 + 8);
          float4 p0 = s0[0], p1 = s0[1], p2 = s1[0], p3 = s1[1];
          f0.x *= p0.x; f0.y *= p0.y; f0.z *= p0.z; f0.w *= p0.w;
          f1.x *= p1.x; f1.y *= p1.y; f1.z *= p1.z; f1.w *= p1.w;
          f2.x *= p2.x; f2.y *= p2.y; f2.z *= p2.z; f2.w *= p2.w;
          f3.x *= p3.x; f3.y *= p3.y; f3.z *= p3.z; f3.w *= p3.w;
        }
        AFrag a;                              // packed f32->bf16 converts
        a.p[0] = f2bf2(f0.x, f0.y);
        a.p[1] = f2bf2(f0.z, f0.w);
        a.p[2] = f2bf2(f1.x, f1.y);
        a.p[3] = f2bf2(f1.z, f1.w);
        a.p[4] = f2bf2(f2.x, f2.y);
        a.p[5] = f2bf2(f2.z, f2.w);
        a.p[6] = f2bf2(f3.x, f3.y);
        a.p[7] = f2bf2(f3.z, f3.w);

        // B fragments: one contiguous 32B LDS read per tn (ds_load_b128 x2)
        const v16bf* bfr = ((const v16bf*)smem) + ks * 128;
        v16bf b0 = bfr[0 * 32 + lane];
        v16bf b1 = bfr[1 * 32 + lane];
        v16bf b2 = bfr[2 * 32 + lane];
        v16bf b3 = bfr[3 * 32 + lane];

        acc[0] = __builtin_amdgcn_wmma_f32_16x16x32_bf16(false, a.v, false, b0,
                                                         (short)0, acc[0], false, false);
        acc[1] = __builtin_amdgcn_wmma_f32_16x16x32_bf16(false, a.v, false, b1,
                                                         (short)0, acc[1], false, false);
        acc[2] = __builtin_amdgcn_wmma_f32_16x16x32_bf16(false, a.v, false, b2,
                                                         (short)0, acc[2], false, false);
        acc[3] = __builtin_amdgcn_wmma_f32_16x16x32_bf16(false, a.v, false, b3,
                                                         (short)0, acc[3], false, false);
      }
    }
    __syncthreads();
  }

  if (active) {
    float post[4];
    if constexpr (SCALE) {
#pragma unroll
      for (int tn = 0; tn < 4; ++tn) post[tn] = postscale[tn * 16 + r];
    }
    const int m0 = stripe * 16 + 8 * h;
#pragma unroll
    for (int tn = 0; tn < 4; ++tn) {
#pragma unroll
      for (int i = 0; i < 8; ++i) {
        float v = acc[tn][i];
        if constexpr (SCALE) v *= post[tn];
        if (doRelu) v = fmaxf(v, 0.f);
        C[(size_t)(m0 + i) * DIM + tn * 16 + r] = v;
      }
    }
  }
}

// out[e] = dot(U[si[e], :64], V[di[e], :64])
__global__ void edge_dot_k(const float* __restrict__ U,
                           const float* __restrict__ V,
                           const int* __restrict__ si,
                           const int* __restrict__ di,
                           float* __restrict__ out, int E) {
  int e = blockIdx.x * blockDim.x + threadIdx.x;
  if (e >= E) return;
  const float4* u = (const float4*)(U + (size_t)si[e] * DIM);
  const float4* v = (const float4*)(V + (size_t)di[e] * DIM);
  float s = 0.f;
#pragma unroll
  for (int i = 0; i < DIM / 4; ++i) {
    float4 a = u[i], b = v[i];
    s = fmaf(a.x, b.x, s);
    s = fmaf(a.y, b.y, s);
    s = fmaf(a.z, b.z, s);
    s = fmaf(a.w, b.w, s);
  }
  out[e] = s;
}

extern "C" void kernel_launch(void* const* d_in, const int* in_sizes, int n_in,
                              void* d_out, int out_size, void* d_ws, size_t ws_size,
                              hipStream_t stream) {
  (void)n_in; (void)out_size; (void)ws_size;
  const float* x_gene = (const float*)d_in[0];
  const float* x_drug = (const float*)d_in[1];
  const int*   ei_gg  = (const int*)d_in[2];
  const int*   gd_src = (const int*)d_in[3];
  const int*   gd_dst = (const int*)d_in[4];
  const int*   ei_dd[4] = {(const int*)d_in[5], (const int*)d_in[6],
                           (const int*)d_in[7], (const int*)d_in[8]};
  const float* W1_gg  = (const float*)d_in[9];
  const float* W1_gd  = (const float*)d_in[10];
  const float* W1_dd[4] = {(const float*)d_in[11], (const float*)d_in[12],
                           (const float*)d_in[13], (const float*)d_in[14]};
  // d_in[15] = W2_gg is dead: layer-2 gene embeddings never reach an output.
  const float* W2_gd  = (const float*)d_in[16];
  const float* W2_dd[4] = {(const float*)d_in[17], (const float*)d_in[18],
                           (const float*)d_in[19], (const float*)d_in[20]};
  const float* M_rel0 = (const float*)d_in[21];
  const float* M_rel1 = (const float*)d_in[22];
  const float* Rm     = (const float*)d_in[23];
  const float* D2     = (const float*)d_in[24];
  const float* D3     = (const float*)d_in[25];

  const long long EGG = in_sizes[2] / 2;
  const long long EGD = in_sizes[3];
  const long long EDD = in_sizes[5] / 2;

  // Workspace layout (floats); layer-2 aggregations reuse layer-1 regions.
  float* ws   = (float*)d_ws;
  float* aggG = ws;                              // NG*FIN
  float* aggD = aggG + (size_t)NG * FIN;         // 5*ND*FIN
  float* zg   = aggD + (size_t)5 * ND * FIN;     // NG*DIM
  float* zd   = zg   + (size_t)NG * DIM;         // ND*DIM
  float* zd2  = zd   + (size_t)ND * DIM;         // ND*DIM
  float* T    = zd2  + (size_t)ND * DIM;         // 4*ND*DIM (T0,T1,P2,P3)
  __bf16* wpk = (__bf16*)(T + (size_t)4 * ND * DIM);   // packed bf16 weights

  // packed weight slots (bf16 elements)
  __bf16* p1_gg = wpk;                                   // 128*64
  __bf16* p1_gd = p1_gg + (size_t)FIN * DIM;
  __bf16* p1_dd[4] = {p1_gd + 1 * (size_t)FIN * DIM, p1_gd + 2 * (size_t)FIN * DIM,
                      p1_gd + 3 * (size_t)FIN * DIM, p1_gd + 4 * (size_t)FIN * DIM};
  __bf16* p2_gd = p1_dd[3] + (size_t)FIN * DIM;          // 64*64 from here on
  __bf16* p2_dd[4] = {p2_gd + 1 * (size_t)DIM * DIM, p2_gd + 2 * (size_t)DIM * DIM,
                      p2_gd + 3 * (size_t)DIM * DIM, p2_gd + 4 * (size_t)DIM * DIM};
  __bf16* pM0 = p2_dd[3] + (size_t)DIM * DIM;
  __bf16* pM1 = pM0 + (size_t)DIM * DIM;
  __bf16* pR  = pM1 + (size_t)DIM * DIM;

  dim3 sblk(256), sgrd(8192);

  // ---- pack all weights to WMMA fragment order (independent of the rest) --
  auto pack = [&](const float* W, __bf16* Wp, int K) {
    int total = (K / 32) * 128;
    pack_w_k<<<dim3((total + 127) / 128), dim3(128), 0, stream>>>(W, Wp, K);
  };
  pack(W1_gg, p1_gg, FIN);
  pack(W1_gd, p1_gd, FIN);
  for (int k = 0; k < 4; ++k) pack(W1_dd[k], p1_dd[k], FIN);
  pack(W2_gd, p2_gd, DIM);
  for (int k = 0; k < 4; ++k) pack(W2_dd[k], p2_dd[k], DIM);
  pack(M_rel0, pM0, DIM);
  pack(M_rel1, pM1, DIM);
  pack(Rm, pR, DIM);

  // ---------------- layer 1: aggregate raw features, then WMMA GEMMs ------
  hipMemsetAsync(aggG, 0, (size_t)NG * FIN * sizeof(float), stream);
  hipMemsetAsync(aggD, 0, (size_t)5 * ND * FIN * sizeof(float), stream);

  scatter_add_k<<<sgrd, sblk, 0, stream>>>(x_gene, ei_gg, ei_gg + EGG, aggG, EGG, 7);
  scatter_add_k<<<sgrd, sblk, 0, stream>>>(x_gene, gd_src, gd_dst, aggD, EGD, 7);
  for (int k = 0; k < 4; ++k)
    scatter_add_k<<<sgrd, sblk, 0, stream>>>(x_drug, ei_dd[k], ei_dd[k] + EDD,
                                             aggD + (size_t)(k + 1) * ND * FIN, EDD, 7);

  const size_t shm128 = (size_t)FIN * DIM * 2;   // 16 KB
  const size_t shm64  = (size_t)DIM * DIM * 2;   //  8 KB

  gemm_wmma_k<FIN, false><<<dim3((NG / 16 + 3) / 4), dim3(128), shm128, stream>>>(
      aggG, nullptr, nullptr, nullptr, nullptr,
      p1_gg, nullptr, nullptr, nullptr, nullptr,
      1, NG, nullptr, nullptr, zg, 1);
  gemm_wmma_k<FIN, false><<<dim3((ND / 16 + 3) / 4), dim3(128), shm128, stream>>>(
      aggD, aggD + (size_t)1 * ND * FIN, aggD + (size_t)2 * ND * FIN,
      aggD + (size_t)3 * ND * FIN, aggD + (size_t)4 * ND * FIN,
      p1_gd, p1_dd[0], p1_dd[1], p1_dd[2], p1_dd[3],
      5, ND, nullptr, nullptr, zd, 1);

  // ---------------- layer 2 (drug embeddings only) -------------------------
  float* agg2_gd = aggG;                       // ND*DIM
  float* agg2_dd = aggD;                       // 4*ND*DIM
  hipMemsetAsync(agg2_gd, 0, (size_t)ND * DIM * sizeof(float), stream);
  hipMemsetAsync(agg2_dd, 0, (size_t)4 * ND * DIM * sizeof(float), stream);

  scatter_add_k<<<sgrd, sblk, 0, stream>>>(zg, gd_src, gd_dst, agg2_gd, EGD, 6);
  for (int k = 0; k < 4; ++k)
    scatter_add_k<<<sgrd, sblk, 0, stream>>>(zd, ei_dd[k], ei_dd[k] + EDD,
                                             agg2_dd + (size_t)k * ND * DIM, EDD, 6);

  gemm_wmma_k<DIM, false><<<dim3((ND / 16 + 3) / 4), dim3(128), shm64, stream>>>(
      agg2_gd, agg2_dd, agg2_dd + (size_t)1 * ND * DIM,
      agg2_dd + (size_t)2 * ND * DIM, agg2_dd + (size_t)3 * ND * DIM,
      p2_gd, p2_dd[0], p2_dd[1], p2_dd[2], p2_dd[3],
      5, ND, nullptr, nullptr, zd2, 1);

  // ---------------- decoders ----------------------------------------------
  float* T0 = T;
  float* T1 = T + (size_t)1 * ND * DIM;
  float* P2 = T + (size_t)2 * ND * DIM;
  float* P3 = T + (size_t)3 * ND * DIM;
  dim3 dgrid((ND / 16 + 3) / 4), dblk(128);
  gemm_wmma_k<DIM, false><<<dgrid, dblk, shm64, stream>>>(
      zd2, nullptr, nullptr, nullptr, nullptr,
      pM0, nullptr, nullptr, nullptr, nullptr, 1, ND, nullptr, nullptr, T0, 0);
  gemm_wmma_k<DIM, false><<<dgrid, dblk, shm64, stream>>>(
      zd2, nullptr, nullptr, nullptr, nullptr,
      pM1, nullptr, nullptr, nullptr, nullptr, 1, ND, nullptr, nullptr, T1, 0);
  gemm_wmma_k<DIM, true><<<dgrid, dblk, shm64, stream>>>(
      zd2, nullptr, nullptr, nullptr, nullptr,
      pR, nullptr, nullptr, nullptr, nullptr, 1, ND, D2, D2, P2, 0);
  gemm_wmma_k<DIM, true><<<dgrid, dblk, shm64, stream>>>(
      zd2, nullptr, nullptr, nullptr, nullptr,
      pR, nullptr, nullptr, nullptr, nullptr, 1, ND, D3, D3, P3, 0);

  float* out = (float*)d_out;
  int eg = (int)((EDD + 255) / 256);
  edge_dot_k<<<eg, 256, 0, stream>>>(T0, zd2, ei_dd[0], ei_dd[0] + EDD, out, (int)EDD);
  edge_dot_k<<<eg, 256, 0, stream>>>(T1, zd2, ei_dd[1], ei_dd[1] + EDD, out + EDD, (int)EDD);
  // dedicom decoders: reference bug uses ei[0] for BOTH src and dst
  edge_dot_k<<<eg, 256, 0, stream>>>(P2, zd2, ei_dd[2], ei_dd[2], out + 2 * EDD, (int)EDD);
  edge_dot_k<<<eg, 256, 0, stream>>>(P3, zd2, ei_dd[3], ei_dd[3], out + 3 * EDD, (int)EDD);
}
